// SSCAM_3487513444489
// MI455X (gfx1250) — matrix operand order repaired and checked by reading, not compiled, not run on hardware
//
#include <hip/hip_runtime.h>

// ---------------------------------------------------------------------------
// Fused SSCAM row-attention kernel for MI455X (gfx1250, wave32, WMMA).
// One workgroup (256 thr = 8 waves) per (n,h) row; all intermediates in LDS
// (exactly 320KB); every GEMM uses v_wmma_f32_16x16x32_f16.
// ---------------------------------------------------------------------------

typedef _Float16 f16;
typedef __attribute__((ext_vector_type(16))) _Float16     v16h;
typedef __attribute__((ext_vector_type(8)))  float        v8f;
typedef __attribute__((ext_vector_type(4)))  unsigned int v4u;

union F16Frag { v16h h; v4u u[2]; unsigned int w[8]; _Float16 e[16]; };
union H2      { unsigned int u; _Float16 h[2]; };

#define CDIM 128
#define WDIM 256
#define HWSZ 65536          // H*W (channel stride in NCHW)
#define NB   4
#define HD   256

// LDS regions (dword offsets), total = 327680 B = 320 KB
#define R_NM   0            // Nm  pair-c  [64][256] dwords (live all phases)
#define R_QA   16384        // Np -> Qm
#define R_QB   32768        // Qp -> Qpm
#define R_V    49152        // Vp -> Vm    pair-v [128][128] dwords
#define R_STG  65536        // 8 waves x 2048 dwords softmax staging
#define LDS_BYTES 327680

// A-operand K-pair index per fragment dword (ISA 16-bit A layout)
__device__ __forceinline__ int pairA(int d, int hl) {
  return (d < 4 ? d : d + 4) + 4 * hl;
}
// bank-conflict-breaking XOR swizzle (consistent across all stores/loads)
__device__ __forceinline__ int swz(int pr, int col) {
  return col ^ (((pr >> 2) & 3) << 4);
}

__device__ __forceinline__ v8f wmma(v16h a, v16h b, v8f c) {
  return __builtin_amdgcn_wmma_f32_16x16x32_f16(false, a, false, b, (short)0, c,
                                                false, false);
}

// A fragment from pair-interleaved LDS matrix (lane holds fixed m = col)
__device__ __forceinline__ v16h ld_afrag(const unsigned int* lds, int col,
                                         int kc, int hl, int rowD, bool doSwz) {
  F16Frag f;
#pragma unroll
  for (int d = 0; d < 8; ++d) {
    int pr = kc * 16 + pairA(d, hl);
    f.w[d] = lds[pr * rowD + (doSwz ? swz(pr, col) : col)];
  }
  return f.h;
}

// B fragment from pair-interleaved LDS matrix (lane holds fixed n = col)
__device__ __forceinline__ v16h ld_bfrag(const unsigned int* lds, int col,
                                         int kc, int hl, int rowD, bool doSwz) {
  F16Frag f;
#pragma unroll
  for (int d = 0; d < 8; ++d) {
    int pr = kc * 16 + pairA(d, hl); // B layout: same K-pair order as A (table order)
    f.w[d] = lds[pr * rowD + (doSwz ? swz(pr, col) : col)];
  }
  return f.h;
}

// Weight A fragment from f16 row-major [128 o][128 c] in global (2 x b128)
__device__ __forceinline__ v16h ld_wfrag(const f16* __restrict__ wg, int m,
                                         int kc, int hl) {
  F16Frag f;
  const f16* p = wg + m * CDIM + kc * 32 + hl * 8;
  f.u[0] = *(const v4u*)p;          // k = base + 0..7
  f.u[1] = *(const v4u*)(p + 16);   // k = base + 16..23
  return f.h;
}

// B fragment straight from global fp32 NCHW slab (on-the-fly f32->f16)
__device__ __forceinline__ v16h ld_bfrag_g(const float* __restrict__ x, int n,
                                           int kc, int hl) {
  F16Frag f;
#pragma unroll
  for (int d = 0; d < 8; ++d) {
    int k0 = kc * 32 + 2 * pairA(d, hl);
    f.e[2 * d]     = (f16)x[(size_t)k0 * HWSZ + n];
    f.e[2 * d + 1] = (f16)x[(size_t)(k0 + 1) * HWSZ + n];
  }
  return f.h;
}

// Out = W(128x128) * In(128x256) + b  ; In from LDS pair-c, Out to LDS pair-c
__device__ __forceinline__ void gemm_q(unsigned int* outLds,
                                       const unsigned int* inLds,
                                       const f16* __restrict__ wg,
                                       const float* __restrict__ bias,
                                       int lane, int ntBeg, int ntEnd) {
  const int hl = lane >> 4, ln = lane & 15;
  for (int nt = ntBeg; nt < ntEnd; ++nt) {
    const int n = nt * 16 + ln;
    v8f acc[8] = {};
#pragma unroll
    for (int kc = 0; kc < 4; ++kc) {
      v16h bf = ld_bfrag(inLds, n, kc, hl, 256, true);
#pragma unroll
      for (int mt = 0; mt < 8; ++mt) {
        v16h af = ld_wfrag(wg, mt * 16 + ln, kc, hl);
        acc[mt] = wmma(af, bf, acc[mt]);
      }
    }
#pragma unroll
    for (int mt = 0; mt < 8; ++mt) {
#pragma unroll
      for (int r = 0; r < 8; r += 2) {       // (m, m+1) pack -> one ds_store_b32
        int m = mt * 16 + 8 * hl + r;
        H2 pk;
        pk.h[0] = (f16)(acc[mt][r]     + bias[m]);
        pk.h[1] = (f16)(acc[mt][r + 1] + bias[m + 1]);
        int pr = m >> 1;
        outLds[pr * 256 + swz(pr, n)] = pk.u;
      }
    }
  }
}

// V = W(128x128) * raw(128x256) + b ; raw from global fp32, Out pair-v [128][128]
__device__ __forceinline__ void gemm_v(unsigned int* outLds,
                                       const float* __restrict__ xg,
                                       const f16* __restrict__ wg,
                                       const float* __restrict__ bias,
                                       int lane, int ntBeg, int ntEnd) {
  _Float16* outH = (_Float16*)outLds;
  const int hl = lane >> 4, ln = lane & 15;
  for (int nt = ntBeg; nt < ntEnd; ++nt) {
    const int n = nt * 16 + ln;              // n = pixel column v
    v8f acc[8] = {};
#pragma unroll
    for (int kc = 0; kc < 4; ++kc) {
      v16h bf = ld_bfrag_g(xg, n, kc, hl);
#pragma unroll
      for (int mt = 0; mt < 8; ++mt) {
        v16h af = ld_wfrag(wg, mt * 16 + ln, kc, hl);
        acc[mt] = wmma(af, bf, acc[mt]);
      }
    }
    const int pr = n >> 1, vlo = n & 1;
#pragma unroll
    for (int mt = 0; mt < 8; ++mt) {
#pragma unroll
      for (int r = 0; r < 8; ++r) {
        int ch = mt * 16 + 8 * hl + r;
        int dw = pr * 128 + swz(pr, ch);
        outH[dw * 2 + vlo] = (f16)(acc[mt][r] + bias[ch]);
      }
    }
  }
}

// One attention path: P = softmax_rows(Qa^T Qb * C^-0.5); out = res + (V P^T)*coef
__device__ __forceinline__ void attn_path(const unsigned int* qa,
                                          const unsigned int* qb,
                                          const unsigned int* vv,
                                          unsigned int* stg,
                                          const float* __restrict__ resid,
                                          const float* __restrict__ coef,
                                          float* __restrict__ outg,
                                          int lane, int wv) {
  const int hl = lane >> 4, ln = lane & 15;
  _Float16* stgH = (_Float16*)stg;
  for (int wi = 0; wi < 2; ++wi) {
    const int wt = wv + 8 * wi;              // this wave's w-tile
    // ---- scores S[w, v] for 16 rows w ----
    v16h A[4];
#pragma unroll
    for (int kc = 0; kc < 4; ++kc)
      A[kc] = ld_afrag(qa, wt * 16 + ln, kc, hl, 256, true);
    v8f S[16];
#pragma unroll
    for (int nt = 0; nt < 16; ++nt) {
      v8f s = {};
#pragma unroll
      for (int kc = 0; kc < 4; ++kc) {
        v16h bf = ld_bfrag(qb, nt * 16 + ln, kc, hl, 256, true);
        s = wmma(A[kc], bf, s);
      }
      S[nt] = s;
    }
    // ---- row softmax (v spans 16 lanes x 16 ntiles; half-waves = rows m+8) ----
    const float sc = 0.08838834764831845f;   // 128^-0.5
#pragma unroll
    for (int r = 0; r < 8; ++r) {
      float mx = -3.0e38f;
#pragma unroll
      for (int nt = 0; nt < 16; ++nt) mx = fmaxf(mx, S[nt][r]);
#pragma unroll
      for (int off = 1; off < 16; off <<= 1) mx = fmaxf(mx, __shfl_xor(mx, off, 32));
      float sum = 0.f;
#pragma unroll
      for (int nt = 0; nt < 16; ++nt) {
        float e = __expf((S[nt][r] - mx) * sc);
        S[nt][r] = e;
        sum += e;
      }
#pragma unroll
      for (int off = 1; off < 16; off <<= 1) sum += __shfl_xor(sum, off, 32);
      float inv = 1.0f / sum;
      int wl = r + 8 * hl;
#pragma unroll
      for (int nt = 0; nt < 16; ++nt) {      // P^T into per-wave stage (pair-v)
        int v = nt * 16 + ln;
        stgH[(v >> 1) * 32 + wl * 2 + (v & 1)] = (f16)(S[nt][r] * inv);
      }
    }
    // ---- apply: F[c, w] = sum_v V[c,v] P[w,v]  (K = v = 256) ----
    v8f acc[8] = {};
#pragma unroll
    for (int kc = 0; kc < 8; ++kc) {
      v16h bf = ld_bfrag(stg, ln, kc, hl, 16, false);
#pragma unroll
      for (int mt = 0; mt < 8; ++mt) {
        v16h af = ld_afrag(vv, mt * 16 + ln, kc, hl, 128, true);
        acc[mt] = wmma(af, bf, acc[mt]);
      }
    }
    // ---- residual + per-channel coefficient, coalesced 64B stores ----
    const int wcol = wt * 16 + ln;
#pragma unroll
    for (int mt = 0; mt < 8; ++mt) {
#pragma unroll
      for (int r = 0; r < 8; ++r) {
        int ch = mt * 16 + 8 * hl + r;
        size_t idx = (size_t)ch * HWSZ + wcol;
        outg[idx] = resid[idx] + acc[mt][r] * coef[ch];
      }
    }
  }
}

// LayerNorm over C for one pixel column (thread t owns column w = t)
__device__ __forceinline__ void layernorm_cols(unsigned int* outLds,
                                               const float* __restrict__ xg,
                                               const float* __restrict__ g,
                                               const float* __restrict__ b,
                                               int w) {
  float s = 0.f, ss = 0.f;
#pragma unroll 4
  for (int c = 0; c < CDIM; ++c) {
    float v = xg[(size_t)c * HWSZ + w];
    s += v;
    ss += v * v;
  }
  float mu   = s * (1.0f / CDIM);
  float var  = ss * (1.0f / CDIM) - mu * mu;
  float rstd = rsqrtf(var + 1e-6f);
  for (int c = 0; c < CDIM; c += 2) {
    float v0 = xg[(size_t)c * HWSZ + w];
    float v1 = xg[(size_t)(c + 1) * HWSZ + w];
    H2 pk;
    pk.h[0] = (f16)((v0 - mu) * rstd * g[c]     + b[c]);
    pk.h[1] = (f16)((v1 - mu) * rstd * g[c + 1] + b[c + 1]);
    int pr = c >> 1;
    outLds[pr * 256 + swz(pr, w)] = pk.u;
  }
}

extern "C" __global__ __launch_bounds__(256)
void sscam_fused(const float* __restrict__ msi, const float* __restrict__ pan,
                 const float* __restrict__ lnw_m, const float* __restrict__ lnb_m,
                 const float* __restrict__ lnw_p, const float* __restrict__ lnb_p,
                 const float* __restrict__ b_mp1, const float* __restrict__ b_pp1,
                 const float* __restrict__ b_mp2, const float* __restrict__ b_pp2,
                 const float* __restrict__ beta, const float* __restrict__ gamma,
                 const f16* __restrict__ wh,    // f16 weights: mp1, pp1, mp2, pp2
                 float* __restrict__ out_msi, float* __restrict__ out_pan) {
  extern __shared__ unsigned int smem[];
  const int row  = blockIdx.x;        // n*H + h
  const int nimg = row >> 8;
  const int h    = row & 255;
  const size_t base = (size_t)nimg * CDIM * HWSZ + (size_t)h * WDIM;
  const float* bm = msi + base;
  const float* bp = pan + base;
  float* om = out_msi + base;
  float* op = out_pan + base;

  const int tid  = threadIdx.x;
  const int lane = tid & 31;
  const int wv   = tid >> 5;

  unsigned int* NM = smem + R_NM;
  unsigned int* QA = smem + R_QA;
  unsigned int* QB = smem + R_QB;
  unsigned int* VV = smem + R_V;
  unsigned int* ST = smem + R_STG + wv * 2048;

  const f16* w_mp1 = wh;
  const f16* w_pp1 = wh + 16384;
  const f16* w_mp2 = wh + 32768;
  const f16* w_pp2 = wh + 49152;

  // Phase 0: LayerNorm both inputs (Np parked in QA region)
  layernorm_cols(NM, bm, lnw_m, lnb_m, tid);
  layernorm_cols(QA, bp, lnw_p, lnb_p, tid);
  __syncthreads();

  // Phase 1: Qp = Wpp1*Np (waves 0-3) | Vp = Wpp2*pan_raw (waves 4-7)
  if (wv < 4) gemm_q(QB, QA, w_pp1, b_pp1, lane, wv * 4, wv * 4 + 4);
  else        gemm_v(VV, bp, w_pp2, b_pp2, lane, (wv - 4) * 4, (wv - 4) * 4 + 4);
  __syncthreads();

  // Phase 2: Qm = Wmp1*Nm -> QA (Np dead)
  gemm_q(QA, NM, w_mp1, b_mp1, lane, wv * 2, wv * 2 + 2);
  __syncthreads();

  // Phase 3: msi' = msi + beta * softmax(Qm^T Qp / sqrt(C)) @ Vp^T
  attn_path(QA, QB, VV, ST, bm, beta, om, lane, wv);
  __syncthreads();

  // Phase 4: Qpm = Wpp1*Nm -> QB (quirk: pan_proj1 of norm_msi) | Vm -> VV
  if (wv < 4) gemm_q(QB, NM, w_pp1, b_pp1, lane, wv * 4, wv * 4 + 4);
  else        gemm_v(VV, bm, w_mp2, b_mp2, lane, (wv - 4) * 4, (wv - 4) * 4 + 4);
  __syncthreads();

  // Phase 5: pan' = pan + gamma * softmax(Qpm^T Qm / sqrt(C)) @ Vm^T
  attn_path(QB, QA, VV, ST, bp, gamma, op, lane, wv);
}

// Convert the four 128x128 fp32 weight matrices to f16 row-major in d_ws
extern "C" __global__ void sscam_wconv(const float* __restrict__ a,
                                       const float* __restrict__ b,
                                       const float* __restrict__ c,
                                       const float* __restrict__ d,
                                       f16* __restrict__ out) {
  int i = blockIdx.x * 256 + threadIdx.x;
  if (i < 16384) {
    out[i]         = (f16)a[i];
    out[i + 16384] = (f16)b[i];
    out[i + 32768] = (f16)c[i];
    out[i + 49152] = (f16)d[i];
  }
}

extern "C" void kernel_launch(void* const* d_in, const int* in_sizes, int n_in,
                              void* d_out, int out_size, void* d_ws, size_t ws_size,
                              hipStream_t stream) {
  (void)in_sizes; (void)n_in; (void)out_size; (void)ws_size;
  const float* msi   = (const float*)d_in[0];
  const float* pan   = (const float*)d_in[1];
  const float* lnwm  = (const float*)d_in[2];
  const float* lnbm  = (const float*)d_in[3];
  const float* lnwp  = (const float*)d_in[4];
  const float* lnbp  = (const float*)d_in[5];
  const float* wmp1  = (const float*)d_in[6];
  const float* bmp1  = (const float*)d_in[7];
  const float* wpp1  = (const float*)d_in[8];
  const float* bpp1  = (const float*)d_in[9];
  const float* wmp2  = (const float*)d_in[10];
  const float* bmp2  = (const float*)d_in[11];
  const float* wpp2  = (const float*)d_in[12];
  const float* bpp2  = (const float*)d_in[13];
  const float* beta  = (const float*)d_in[14];
  const float* gamma = (const float*)d_in[15];

  f16*   wh      = (f16*)d_ws;
  float* out_msi = (float*)d_out;
  float* out_pan = out_msi + (size_t)NB * CDIM * HWSZ;

  sscam_wconv<<<64, 256, 0, stream>>>(wmp1, wpp1, wmp2, wpp2, wh);
  sscam_fused<<<NB * HD, 256, LDS_BYTES, stream>>>(
      msi, pan, lnwm, lnbm, lnwp, lnbp, bmp1, bpp1, bmp2, bpp2, beta, gamma, wh,
      out_msi, out_pan);
}
